// RNN_73418170958322
// MI455X (gfx1250) — matrix operand order repaired and checked by reading, not compiled, not run on hardware
//
#include <hip/hip_runtime.h>
#include <math.h>

#define DEVINL __device__ __forceinline__

typedef __bf16 bf16_t;
typedef __attribute__((ext_vector_type(8)))  __bf16 bf16x8;
typedef __attribute__((ext_vector_type(16))) __bf16 bf16x16;
typedef __attribute__((ext_vector_type(8)))  float  f32x8;

constexpr int T_ = 1024, B_ = 128, V_ = 512, H_ = 2048, O_ = 512;
constexpr int NWG  = 32;    // persistent workgroups (must all be co-resident)
constexpr int NTHR = 256;   // 8 wave32s

// ---- workspace layout (bytes) ----
constexpr size_t WH_OFF   = 0;                                  // Wh  bf16 [H,H]
constexpr size_t WI_OFF   = WH_OFF   + (size_t)H_ * H_ * 2;     // Wi  bf16 [H,V]
constexpr size_t WO_OFF   = WI_OFF   + (size_t)H_ * V_ * 2;     // Wo  bf16 [O,H]
constexpr size_t EMB_OFF  = WO_OFF   + (size_t)O_ * H_ * 2;     // emb bf16 [V,V]
constexpr size_t HBUF_OFF = EMB_OFF  + (size_t)V_ * V_ * 2;     // h double buffer bf16 [2,B,H]
constexpr size_t HB_OFF   = HBUF_OFF + (size_t)2 * B_ * H_ * 2; // bi+bh f32 [H]
constexpr size_t CNT_OFF  = HB_OFF   + (size_t)H_ * 4;          // barrier {count, gen}

DEVINL bf16_t f2bf(float f) {
  unsigned u = __builtin_bit_cast(unsigned, f);
  u += 0x7FFFu + ((u >> 16) & 1u);                 // round-to-nearest-even
  unsigned short s = (unsigned short)(u >> 16);
  return __builtin_bit_cast(bf16_t, s);
}
DEVINL float bf2f(bf16_t b) {
  unsigned u = ((unsigned)__builtin_bit_cast(unsigned short, b)) << 16;
  return __builtin_bit_cast(float, u);
}

DEVINL bf16x16 cat8(bf16x8 lo, bf16x8 hi) {
  return __builtin_shufflevector(lo, hi, 0,1,2,3,4,5,6,7,8,9,10,11,12,13,14,15);
}
// A fragment 16x32 bf16. rowp = per-lane row base (M = lane&15). ISA layout:
// lanes 0-15 hold K = k0+0..7 / k0+16..23, lanes 16-31 hold K = k0+8..15 / k0+24..31.
DEVINL bf16x16 load_a(const bf16_t* rowp, int k0, int lh) {
  bf16x8 lo = *(const bf16x8*)(rowp + k0 + lh * 8);
  bf16x8 hi = *(const bf16x8*)(rowp + k0 + 16 + lh * 8);
  return cat8(lo, hi);
}
// B fragment 32x16 where B[k][n] = W[n][k] (W row-major). rowp = W + n*ld,
// n = lane&15. Lanes 0-15 hold K = k0+0..7 / k0+16..23, lanes 16-31 hold K = k0+8..15 / k0+24..31.
DEVINL bf16x16 load_b(const bf16_t* rowp, int k0, int lh) {
  bf16x8 lo = *(const bf16x8*)(rowp + k0 + lh * 8);
  bf16x8 hi = *(const bf16x8*)(rowp + k0 + 16 + lh * 8);
  return cat8(lo, hi);
}
DEVINL f32x8 wmma_bf16(bf16x16 a, bf16x16 b, f32x8 c) {
  return __builtin_amdgcn_wmma_f32_16x16x32_bf16(false, a, false, b, (short)0, c,
                                                 false, false);
}

// Grid-wide barrier: monotone generation counter + arrival count, per-block epoch.
DEVINL void grid_barrier(unsigned* cnt, unsigned* gen, unsigned epoch) {
  __syncthreads();
  if (threadIdx.x == 0) {
    __threadfence();                               // release stores to device scope
    if (atomicAdd(cnt, 1u) == (unsigned)(NWG - 1)) {
      atomicSub(cnt, (unsigned)NWG);               // reset for next epoch
      __threadfence();
      atomicMax(gen, epoch);                       // open the gate
    } else {
      while (atomicAdd(gen, 0u) < epoch) __builtin_amdgcn_s_sleep(1);
    }
  }
  __syncthreads();
  __threadfence();                                 // acquire: invalidate stale lines
}

__global__ void cvt_f32_bf16(const float* __restrict__ s, bf16_t* __restrict__ d, int n) {
  for (int i = blockIdx.x * blockDim.x + threadIdx.x; i < n; i += gridDim.x * blockDim.x)
    d[i] = f2bf(s[i]);
}

__global__ void init_state(const float* __restrict__ bi, const float* __restrict__ bh,
                           float* __restrict__ hb, bf16_t* __restrict__ h0,
                           unsigned* __restrict__ cnt) {
  int i = blockIdx.x * blockDim.x + threadIdx.x;
  int n = gridDim.x * blockDim.x;
  for (int j = i; j < H_; j += n) hb[j] = bi[j] + bh[j];
  for (int j = i; j < B_ * H_; j += n) h0[j] = f2bf(0.0f);
  if (i == 0) { cnt[0] = 0u; cnt[1] = 0u; }
}

__global__ __launch_bounds__(NTHR) void rnn_persistent(
    const int* __restrict__ x, const float* __restrict__ bo,
    const float* __restrict__ hb,
    const bf16_t* __restrict__ embb, const bf16_t* __restrict__ wib,
    const bf16_t* __restrict__ whb,  const bf16_t* __restrict__ wob,
    bf16_t* __restrict__ hbuf, float* __restrict__ out,
    unsigned* cnt, unsigned* gen)
{
  const int tid  = threadIdx.x;
  const int wg   = blockIdx.x;       // 0..31
  const int wave = tid >> 5;         // 0..7
  const int lane = tid & 31;
  const int lm   = lane & 15;        // tile row/col within fragment
  const int lh   = lane >> 4;        // low/high lane half

  // Phase A: 8 M-blocks x (4 col groups x 8 waves x 4 tiles) covers [128 x 2048]
  const int m0A = (wg >> 2) << 4;
  const int n0A = (wg & 3) * 512 + wave * 64;

  // Phase B: 256 waves -> 8 x 32 tiles covers [128 x 512]
  const int WID = wg * 8 + wave;
  const int m0B = (WID >> 5) << 4;
  const int n0B = (WID & 31) << 4;

  // Per-wave invariant row pointers (K-offsets applied inside the loops)
  const bf16_t* b0v = wib + (size_t)(n0A +  0 + lm) * V_;
  const bf16_t* b1v = wib + (size_t)(n0A + 16 + lm) * V_;
  const bf16_t* b2v = wib + (size_t)(n0A + 32 + lm) * V_;
  const bf16_t* b3v = wib + (size_t)(n0A + 48 + lm) * V_;
  const bf16_t* b0h = whb + (size_t)(n0A +  0 + lm) * H_;
  const bf16_t* b1h = whb + (size_t)(n0A + 16 + lm) * H_;
  const bf16_t* b2h = whb + (size_t)(n0A + 32 + lm) * H_;
  const bf16_t* b3h = whb + (size_t)(n0A + 48 + lm) * H_;
  const bf16_t* brow_o = wob + (size_t)(n0B + lm) * H_;
  const float g0 = hb[n0A +  0 + lm];
  const float g1 = hb[n0A + 16 + lm];
  const float g2 = hb[n0A + 32 + lm];
  const float g3 = hb[n0A + 48 + lm];
  const float ob = bo[n0B + lm];
  const int   mrowA = m0A + lm;

  // hNew(t) = tanh(e_t @ Wi^T + hOld @ Wh^T + (bi+bh)), hNew = buf[(t+1)&1]
  auto phaseA = [&](int t) {
    const bf16_t* hOld = hbuf + (size_t)(t & 1) * (B_ * H_);
    bf16_t*       hNew = hbuf + (size_t)((t + 1) & 1) * (B_ * H_);

    f32x8 c0 = {}, c1 = {}, c2 = {}, c3 = {};
    // fused embedding gather: per-lane row of emb selected by token id
    const bf16_t* arow_e = embb + (size_t)x[(size_t)t * B_ + mrowA] * V_;
    #pragma unroll 4
    for (int k0 = 0; k0 < V_; k0 += 32) {
      bf16x16 a = load_a(arow_e, k0, lh);
      c0 = wmma_bf16(a, load_b(b0v, k0, lh), c0);
      c1 = wmma_bf16(a, load_b(b1v, k0, lh), c1);
      c2 = wmma_bf16(a, load_b(b2v, k0, lh), c2);
      c3 = wmma_bf16(a, load_b(b3v, k0, lh), c3);
    }
    const bf16_t* arow_h = hOld + (size_t)mrowA * H_;
    #pragma unroll 4
    for (int k0 = 0; k0 < H_; k0 += 32) {
      bf16x16 a = load_a(arow_h, k0, lh);
      c0 = wmma_bf16(a, load_b(b0h, k0, lh), c0);
      c1 = wmma_bf16(a, load_b(b1h, k0, lh), c1);
      c2 = wmma_bf16(a, load_b(b2h, k0, lh), c2);
      c3 = wmma_bf16(a, load_b(b3h, k0, lh), c3);
    }
    #pragma unroll
    for (int v = 0; v < 8; ++v) {                  // C/D layout: M = v + 8*lh, N = lm
      bf16_t* rp = hNew + (size_t)(m0A + v + 8 * lh) * H_;
      rp[n0A +  0 + lm] = f2bf(tanhf(c0[v] + g0));
      rp[n0A + 16 + lm] = f2bf(tanhf(c1[v] + g1));
      rp[n0A + 32 + lm] = f2bf(tanhf(c2[v] + g2));
      rp[n0A + 48 + lm] = f2bf(tanhf(c3[v] + g3));
    }
  };

  // out_t = hNew(t) @ Wo^T + bo   (reads buf[(t+1)&1])
  auto phaseB = [&](int t) {
    f32x8 oc = {};
    const bf16_t* hNew = hbuf + (size_t)((t + 1) & 1) * (B_ * H_);
    const bf16_t* arow_o = hNew + (size_t)(m0B + lm) * H_;
    #pragma unroll 4
    for (int k0 = 0; k0 < H_; k0 += 32)
      oc = wmma_bf16(load_a(arow_o, k0, lh), load_b(brow_o, k0, lh), oc);
    float* orow = out + (size_t)t * B_ * O_;
    #pragma unroll
    for (int v = 0; v < 8; ++v)
      orow[(size_t)(m0B + v + 8 * lh) * O_ + n0B + lm] = oc[v] + ob;
  };

  // Software pipeline: one grid barrier per timestep.
  //   A(0); for t: { bar; A(t+1); B(t); } bar
  // Safety: A(t+1) writes buf[t&1]; its prior readers A(t) and B(t-1) are both
  // program-order-before bar(t) in every workgroup (double buffer => no WAR).
  unsigned epoch = 0;
  phaseA(0);
  for (int t = 0; t < T_; ++t) {
    grid_barrier(cnt, gen, ++epoch);
    if (t + 1 < T_) phaseA(t + 1);                 // critical path first
    phaseB(t);
  }
  grid_barrier(cnt, gen, ++epoch);

  // ---------- final hidden state (second tuple output, f32) ----------
  const bf16_t* hF = hbuf + (size_t)(T_ & 1) * (B_ * H_);
  float* hout = out + (size_t)T_ * B_ * O_;
  for (int i = wg * NTHR + tid; i < B_ * H_; i += NWG * NTHR)
    hout[i] = bf2f(hF[i]);
}

extern "C" void kernel_launch(void* const* d_in, const int* in_sizes, int n_in,
                              void* d_out, int out_size, void* d_ws, size_t ws_size,
                              hipStream_t stream) {
  const int*   x   = (const int*)  d_in[0];
  const float* emb = (const float*)d_in[1];
  const float* Wi  = (const float*)d_in[2];
  const float* bi  = (const float*)d_in[3];
  const float* Wh  = (const float*)d_in[4];
  const float* bh  = (const float*)d_in[5];
  const float* Wo  = (const float*)d_in[6];
  const float* bo  = (const float*)d_in[7];
  float* out = (float*)d_out;

  char* ws = (char*)d_ws;
  bf16_t*   whb  = (bf16_t*)(ws + WH_OFF);
  bf16_t*   wib  = (bf16_t*)(ws + WI_OFF);
  bf16_t*   wob  = (bf16_t*)(ws + WO_OFF);
  bf16_t*   embb = (bf16_t*)(ws + EMB_OFF);
  bf16_t*   hbuf = (bf16_t*)(ws + HBUF_OFF);
  float*    hb   = (float*)  (ws + HB_OFF);
  unsigned* cnt  = (unsigned*)(ws + CNT_OFF);

  cvt_f32_bf16<<<512, 256, 0, stream>>>(Wh,  whb,  H_ * H_);
  cvt_f32_bf16<<<256, 256, 0, stream>>>(Wi,  wib,  H_ * V_);
  cvt_f32_bf16<<<256, 256, 0, stream>>>(Wo,  wob,  O_ * H_);
  cvt_f32_bf16<<<128, 256, 0, stream>>>(emb, embb, V_ * V_);
  init_state  <<<128, 256, 0, stream>>>(bi, bh, hb, hbuf, cnt);

  rnn_persistent<<<NWG, NTHR, 0, stream>>>(x, bo, hb, embb, wib, whb, wob,
                                           hbuf, out, cnt, cnt + 1);
}
